// SpikeQwen3Attention_43860206027297
// MI455X (gfx1250) — matrix operand order repaired and checked by reading, not compiled, not run on hardware
//
#include <hip/hip_runtime.h>
#include <stdint.h>

// ---------------------------------------------------------------------------
// Qwen3-style attention block for MI455X (gfx1250, wave32, WMMA).
// Pipeline: [QKV GEMM + RMSNorm + RoPE] -> [flash attention] -> [out proj].
// All matrix math via v_wmma_f32_16x16x32_bf16 (f32 accumulate).
// Async global->LDS staging (ASYNCcnt path) where the copy is transform-free.
// ---------------------------------------------------------------------------

typedef __bf16 bf16;
typedef __attribute__((ext_vector_type(16))) __bf16 v16bf;
typedef __attribute__((ext_vector_type(8)))  float  v8f;
typedef __attribute__((ext_vector_type(4)))  unsigned int u32x4;
typedef __attribute__((ext_vector_type(2)))  unsigned int u32x2;
typedef __attribute__((ext_vector_type(4)))  float  f32x4;

union Frag { v16bf v; u32x4 q[2]; };

#define WMMA_BF16(A,B,C) __builtin_amdgcn_wmma_f32_16x16x32_bf16(false,(A),false,(B),(short)0,(C),false,false)

// --- optional CDNA5 async global->LDS path (guarded; falls back cleanly) ---
#if defined(__has_builtin)
#  if __has_builtin(__builtin_amdgcn_global_load_async_to_lds_b128) && \
      __has_builtin(__builtin_amdgcn_s_wait_asynccnt)
#    define HAVE_ASYNC_LDS 1
#  endif
#endif
#ifndef HAVE_ASYNC_LDS
#  define HAVE_ASYNC_LDS 0
#endif

#define ASG __attribute__((address_space(1)))
#define ASL __attribute__((address_space(3)))
typedef int v4i_g __attribute__((__vector_size__(16)));   // matches builtin proto

#if HAVE_ASYNC_LDS
#define GLOBAL_ASYNC_B128(gsrc, ldst)                                     \
  __builtin_amdgcn_global_load_async_to_lds_b128(                         \
      (ASG v4i_g*)(gsrc), (ASL v4i_g*)(ldst), 0, 0)
#endif

constexpr int Bx = 2, Sx = 2048, Dx = 2048, HQx = 16, HKx = 8, HDx = 128;
constexpr float EPSx   = 1e-6f;
constexpr float SCALEx = 0.08838834764831845f;          // 128^-0.5
constexpr float L2T_64 = 0.20762050593045951f;          // log2(10000)/64

static __device__ __forceinline__ unsigned pack_bf16x2(float a, float b) {
  bf16 ha = (bf16)a, hb = (bf16)b;
  unsigned short ua, ub;
  __builtin_memcpy(&ua, &ha, 2);
  __builtin_memcpy(&ub, &hb, 2);
  return (unsigned)ua | ((unsigned)ub << 16);
}

// ---------------------------------------------------------------------------
// Kernel A: X[4096,2048] @ [Wq|Wk|Wv] -> per-head tiles, fused RMSNorm + RoPE.
// grid: (32 n-blocks over 4096 concat cols, 32 m-blocks over B*S). 256 thr.
// ---------------------------------------------------------------------------
__global__ __launch_bounds__(256) void qkv_kernel(
    const float* __restrict__ X,  const float* __restrict__ Wq,
    const float* __restrict__ Wk, const float* __restrict__ Wv,
    const float* __restrict__ qnw, const float* __restrict__ knw,
    bf16* __restrict__ Qb, bf16* __restrict__ Kb, bf16* __restrict__ Vb)
{
  constexpr int LDA = 40, LDB = 40, LDC = 132;
  __shared__ bf16  As[128 * LDA];       // A tile  (bf16, [row][k])
  __shared__ bf16  Bt[128 * LDB];       // W tile  (bf16, transposed [n][k])
  __shared__ bf16  CT[128 * LDC];       // normalized tile for RoPE pass
  __shared__ float partial[4][128];
  __shared__ float rinv[128];
  __shared__ float wnorm[128];

  const int t  = threadIdx.x;
  const int nb = blockIdx.x;
  const int mb = blockIdx.y;
  const int rowbase = mb * 128;

  const float* W; int ncb, Lw, mode, head;     // mode: 0=Q 1=K 2=V
  if (nb < 16)      { W = Wq; ncb = nb * 128;        Lw = HQx * HDx; mode = 0; head = nb; }
  else if (nb < 24) { W = Wk; ncb = (nb - 16) * 128; Lw = HKx * HDx; mode = 1; head = nb - 16; }
  else              { W = Wv; ncb = (nb - 24) * 128; Lw = HKx * HDx; mode = 2; head = nb - 24; }

  if (t < 128) wnorm[t] = (mode == 0) ? qnw[t] : (mode == 1 ? knw[t] : 1.0f);

  const int wv = t >> 5, lane = t & 31;
  const int wm = wv >> 2, wn = wv & 3;          // 2x4 wave grid, 64x32 per wave
  const int l16 = lane & 15, lh = lane >> 4;

  const v8f vzero = {};
  v8f acc[4][2];
  #pragma unroll
  for (int i = 0; i < 4; i++)
    #pragma unroll
    for (int j = 0; j < 2; j++) acc[i][j] = vzero;

  for (int k0 = 0; k0 < Dx; k0 += 32) {
    __syncthreads();
    // stage A: 128x32 fp32 -> bf16
    #pragma unroll
    for (int i = 0; i < 4; i++) {
      int slot = t + i * 256;
      int r = slot >> 3, c4 = slot & 7;
      const f32x4 x4 = *(const f32x4*)(X + (size_t)(rowbase + r) * Dx + k0 + c4 * 4);
      u32x2 p; p.x = pack_bf16x2(x4.x, x4.y); p.y = pack_bf16x2(x4.z, x4.w);
      *(u32x2*)(&As[r * LDA + c4 * 4]) = p;
    }
    // stage W: 32x128 fp32 -> bf16 transposed [n][k]
    #pragma unroll
    for (int i = 0; i < 4; i++) {
      int slot = t + i * 256;
      int kr = slot >> 5, c4 = slot & 31;
      const float* wp = W + (size_t)(k0 + kr) * Lw + ncb + c4 * 4;
      if (k0 + 32 < Dx) __builtin_prefetch(wp + (size_t)32 * Lw, 0, 0);  // global_prefetch_b8
      const f32x4 w4 = *(const f32x4*)wp;
      int cb = c4 * 4;
      Bt[(cb + 0) * LDB + kr] = (bf16)w4.x;
      Bt[(cb + 1) * LDB + kr] = (bf16)w4.y;
      Bt[(cb + 2) * LDB + kr] = (bf16)w4.z;
      Bt[(cb + 3) * LDB + kr] = (bf16)w4.w;
    }
    __syncthreads();

    Frag a[4], b[2];
    #pragma unroll
    for (int tm = 0; tm < 4; tm++) {
      int row = wm * 64 + tm * 16 + l16;
      int kb0 = lh * 8;
      a[tm].q[0] = *(const u32x4*)(&As[row * LDA + kb0]);
      a[tm].q[1] = *(const u32x4*)(&As[row * LDA + kb0 + 16]);
    }
    #pragma unroll
    for (int tn = 0; tn < 2; tn++) {
      int col = wn * 32 + tn * 16 + l16;
      int kb  = lh * 8;
      b[tn].q[0] = *(const u32x4*)(&Bt[col * LDB + kb]);
      b[tn].q[1] = *(const u32x4*)(&Bt[col * LDB + kb + 16]);
    }
    #pragma unroll
    for (int tm = 0; tm < 4; tm++)
      #pragma unroll
      for (int tn = 0; tn < 2; tn++)
        acc[tm][tn] = WMMA_BF16(a[tm].v, b[tn].v, acc[tm][tn]);
  }

  if (mode == 2) {   // V: straight bf16 store, [B,HK,S,HD]
    #pragma unroll
    for (int tm = 0; tm < 4; tm++)
      #pragma unroll
      for (int tn = 0; tn < 2; tn++)
        #pragma unroll
        for (int r = 0; r < 8; r++) {
          int rt = wm * 64 + tm * 16 + r + lh * 8;
          int ct = wn * 32 + tn * 16 + l16;
          int gr = rowbase + rt;
          int bb = gr / Sx, ss = gr % Sx;
          Vb[(((size_t)bb * HKx + head) * Sx + ss) * HDx + ct] = (bf16)acc[tm][tn][r];
        }
    return;
  }

  // RMSNorm: per-row sum of squares (shfl within 16-lane row groups + LDS)
  #pragma unroll
  for (int tm = 0; tm < 4; tm++)
    #pragma unroll
    for (int r = 0; r < 8; r++) {
      float ss = acc[tm][0][r] * acc[tm][0][r] + acc[tm][1][r] * acc[tm][1][r];
      ss += __shfl_xor(ss, 1); ss += __shfl_xor(ss, 2);
      ss += __shfl_xor(ss, 4); ss += __shfl_xor(ss, 8);
      if (l16 == 0) partial[wn][wm * 64 + tm * 16 + r + lh * 8] = ss;
    }
  __syncthreads();
  if (t < 128) {
    float s = partial[0][t] + partial[1][t] + partial[2][t] + partial[3][t];
    rinv[t] = rsqrtf(s * (1.0f / 128.0f) + EPSx);
  }
  __syncthreads();
  #pragma unroll
  for (int tm = 0; tm < 4; tm++)
    #pragma unroll
    for (int tn = 0; tn < 2; tn++)
      #pragma unroll
      for (int r = 0; r < 8; r++) {
        int rt = wm * 64 + tm * 16 + r + lh * 8;
        int ct = wn * 32 + tn * 16 + l16;
        CT[rt * LDC + ct] = (bf16)(acc[tm][tn][r] * rinv[rt] * wnorm[ct]);
      }
  __syncthreads();

  // RoPE + store (one thread per token row); SCALE folded into Q
  if (t < 128) {
    int gr = rowbase + t;
    int bb = gr / Sx, ss = gr % Sx;
    float pos = (float)ss;
    bf16* dst = (mode == 0)
        ? (Qb + (((size_t)bb * HQx + head) * Sx + ss) * HDx)
        : (Kb + (((size_t)bb * HKx + head) * Sx + ss) * HDx);
    float mul = (mode == 0) ? SCALEx : 1.0f;
    for (int j = 0; j < 64; j++) {
      float ang = pos * exp2f(-(float)j * L2T_64);
      float sn, cs; __sincosf(ang, &sn, &cs);
      float x1 = (float)CT[t * LDC + j];
      float x2 = (float)CT[t * LDC + j + 64];
      dst[j]      = (bf16)((x1 * cs - x2 * sn) * mul);
      dst[j + 64] = (bf16)((x2 * cs + x1 * sn) * mul);
    }
  }
}

// ---------------------------------------------------------------------------
// Kernel B: causal GQA flash attention. grid: (B*HQ, S/128). 256 thr (8 waves);
// each wave owns 16 query rows, all waves share K/V blocks of 64 keys in LDS.
// K block staged via GLOBAL_LOAD_ASYNC_TO_LDS_B128 when available.
// ---------------------------------------------------------------------------
__global__ __launch_bounds__(256) void attn_kernel(
    const bf16* __restrict__ Qb, const bf16* __restrict__ Kb,
    const bf16* __restrict__ Vb, bf16* __restrict__ Ctx)
{
  constexpr int LKS = 136, LVT = 72, LP = 72;
  __shared__ bf16 Ks[64 * LKS];      // K block  [tok][d]
  __shared__ bf16 Vt[128 * LVT];     // V block transposed [d][tok]
  __shared__ bf16 Pls[8][16 * LP];   // per-wave P staging [qrow][tok]

  const int t = threadIdx.x, wv = t >> 5, lane = t & 31;
  const int l16 = lane & 15, lh = lane >> 4;
  const int bh = blockIdx.x, b = bh / HQx, h = bh % HQx;
  const int hk = h / (HQx / HKx);
  const int qblk = blockIdx.y, qbase = qblk * 128;

  const bf16* Qp = Qb + ((size_t)b * HQx + h)  * Sx * HDx;
  const bf16* Kp = Kb + ((size_t)b * HKx + hk) * Sx * HDx;
  const bf16* Vp = Vb + ((size_t)b * HKx + hk) * Sx * HDx;

  // Q fragments for this wave's 16 rows (4 d-chunks of 32)
  Frag qf[4];
  {
    int srow = qbase + wv * 16 + l16;
    int kb0 = lh * 8;
    #pragma unroll
    for (int dc = 0; dc < 4; dc++) {
      qf[dc].q[0] = *(const u32x4*)(Qp + (size_t)srow * HDx + dc * 32 + kb0);
      qf[dc].q[1] = *(const u32x4*)(Qp + (size_t)srow * HDx + dc * 32 + kb0 + 16);
    }
  }

  const v8f vzero = {};
  v8f o[8]; float mrow[8], lrow[8];
  #pragma unroll
  for (int i = 0; i < 8; i++) { o[i] = vzero; mrow[i] = -1e30f; lrow[i] = 0.0f; }

  const int nkb = qblk * 2 + 2;              // cover keys up to qbase+127
  for (int kb = 0; kb < nkb; kb++) {
    __syncthreads();
    // K block -> LDS
#if HAVE_ASYNC_LDS
    #pragma unroll
    for (int i = 0; i < 4; i++) {
      int slot = t + i * 256;
      int kr = slot >> 4, c8 = slot & 15;
      GLOBAL_ASYNC_B128(Kp + (size_t)(kb * 64 + kr) * HDx + c8 * 8,
                        &Ks[kr * LKS + c8 * 8]);
    }
#else
    #pragma unroll
    for (int i = 0; i < 4; i++) {
      int slot = t + i * 256;
      int kr = slot >> 4, c8 = slot & 15;
      u32x4 d = *(const u32x4*)(Kp + (size_t)(kb * 64 + kr) * HDx + c8 * 8);
      *(u32x4*)(&Ks[kr * LKS + c8 * 8]) = d;
    }
#endif
    // V block -> LDS transposed (per-element transform, stays on manual path)
    #pragma unroll
    for (int i = 0; i < 4; i++) {
      int slot = t + i * 256;
      int kr = slot >> 4, c8 = slot & 15;
      union { u32x4 q; bf16 hh[8]; } u;
      u.q = *(const u32x4*)(Vp + (size_t)(kb * 64 + kr) * HDx + c8 * 8);
      #pragma unroll
      for (int j = 0; j < 8; j++) Vt[(c8 * 8 + j) * LVT + kr] = u.hh[j];
    }
#if HAVE_ASYNC_LDS
    __builtin_amdgcn_s_wait_asynccnt(0);   // own async loads done before barrier
#endif
    __syncthreads();

    // scores: 16 q x 64 k  (4 WMMA tiles, each sum of 4 d-chunks)
    v8f st[4];
    #pragma unroll
    for (int ks = 0; ks < 4; ks++) {
      v8f s = vzero;
      int tok = ks * 16 + l16;
      int kbb = lh * 8;
      #pragma unroll
      for (int dc = 0; dc < 4; dc++) {
        Frag kf;
        kf.q[0] = *(const u32x4*)(&Ks[tok * LKS + dc * 32 + kbb]);
        kf.q[1] = *(const u32x4*)(&Ks[tok * LKS + dc * 32 + kbb + 16]);
        s = WMMA_BF16(qf[dc].v, kf.v, s);
      }
      st[ks] = s;
    }

    // causal mask + online softmax (rows replicated across 16-lane groups)
    float mnew[8];
    #pragma unroll
    for (int r = 0; r < 8; r++) {
      int tq = qbase + wv * 16 + r + lh * 8;
      float mx = mrow[r];
      #pragma unroll
      for (int ks = 0; ks < 4; ks++) {
        int tk = kb * 64 + ks * 16 + l16;
        float val = (tk <= tq) ? st[ks][r] : -1e30f;
        st[ks][r] = val;
        mx = fmaxf(mx, val);
      }
      mx = fmaxf(mx, __shfl_xor(mx, 1));
      mx = fmaxf(mx, __shfl_xor(mx, 2));
      mx = fmaxf(mx, __shfl_xor(mx, 4));
      mx = fmaxf(mx, __shfl_xor(mx, 8));
      mnew[r] = mx;
    }
    #pragma unroll
    for (int r = 0; r < 8; r++) {
      float scl = __expf(mrow[r] - mnew[r]);
      float rs = 0.0f;
      #pragma unroll
      for (int ks = 0; ks < 4; ks++) {
        float p = __expf(st[ks][r] - mnew[r]);
        st[ks][r] = p; rs += p;
      }
      rs += __shfl_xor(rs, 1); rs += __shfl_xor(rs, 2);
      rs += __shfl_xor(rs, 4); rs += __shfl_xor(rs, 8);
      lrow[r] = lrow[r] * scl + rs;
      mrow[r] = mnew[r];
      #pragma unroll
      for (int dt = 0; dt < 8; dt++) o[dt][r] *= scl;
    }

    // restage P (C layout -> A-fragment layout) via LDS
    #pragma unroll
    for (int ks = 0; ks < 4; ks++)
      #pragma unroll
      for (int r = 0; r < 8; r++)
        Pls[wv][(r + lh * 8) * LP + ks * 16 + l16] = (bf16)st[ks][r];
    __syncthreads();

    // PV: o[16 x 128] += P[16 x 64] @ V[64 x 128]
    #pragma unroll
    for (int kh = 0; kh < 2; kh++) {
      Frag pf;
      int kb0 = lh * 8;
      pf.q[0] = *(const u32x4*)(&Pls[wv][l16 * LP + kh * 32 + kb0]);
      pf.q[1] = *(const u32x4*)(&Pls[wv][l16 * LP + kh * 32 + kb0 + 16]);
      #pragma unroll
      for (int dt = 0; dt < 8; dt++) {
        Frag vf;
        int d = dt * 16 + l16;
        vf.q[0] = *(const u32x4*)(&Vt[d * LVT + kh * 32 + lh * 8]);
        vf.q[1] = *(const u32x4*)(&Vt[d * LVT + kh * 32 + lh * 8 + 16]);
        o[dt] = WMMA_BF16(pf.v, vf.v, o[dt]);
      }
    }
  }

  // epilogue: normalize, write context [B,S,HQ*HD] bf16
  bf16* Cp = Ctx + (size_t)b * Sx * (HQx * HDx) + (size_t)h * HDx;
  #pragma unroll
  for (int r = 0; r < 8; r++) {
    float inv = 1.0f / lrow[r];
    int tq = qbase + wv * 16 + r + lh * 8;
    #pragma unroll
    for (int dt = 0; dt < 8; dt++)
      Cp[(size_t)tq * (HQx * HDx) + dt * 16 + l16] = (bf16)(o[dt][r] * inv);
  }
}

// ---------------------------------------------------------------------------
// Kernel C: ctx[4096,2048]bf16 @ Wo[2048,2048]fp32 -> out fp32.
// grid: (16 n-blocks, 32 m-blocks). 256 thr. A tile staged async when possible.
// ---------------------------------------------------------------------------
__global__ __launch_bounds__(256) void oproj_kernel(
    const bf16* __restrict__ Ctx, const float* __restrict__ Wo,
    float* __restrict__ Out)
{
  constexpr int LDA = 40, LDB = 40;
  __shared__ bf16 As[128 * LDA];
  __shared__ bf16 Bt[128 * LDB];

  const int t = threadIdx.x, wv = t >> 5, lane = t & 31;
  const int l16 = lane & 15, lh = lane >> 4;
  const int wm = wv >> 2, wn = wv & 3;
  const int nb = blockIdx.x, mb = blockIdx.y;
  const int rowbase = mb * 128, ncb = nb * 128;
  constexpr int N = HQx * HDx;   // 2048

  const v8f vzero = {};
  v8f acc[4][2];
  #pragma unroll
  for (int i = 0; i < 4; i++)
    #pragma unroll
    for (int j = 0; j < 2; j++) acc[i][j] = vzero;

  for (int k0 = 0; k0 < N; k0 += 32) {
    __syncthreads();
    // A tile (already bf16): plain copy -> async LDS path when available
#if HAVE_ASYNC_LDS
    #pragma unroll
    for (int i = 0; i < 2; i++) {
      int slot = t + i * 256;
      int r = slot >> 2, c8 = slot & 3;
      GLOBAL_ASYNC_B128(Ctx + (size_t)(rowbase + r) * N + k0 + c8 * 8,
                        &As[r * LDA + c8 * 8]);
    }
#else
    #pragma unroll
    for (int i = 0; i < 2; i++) {
      int slot = t + i * 256;
      int r = slot >> 2, c8 = slot & 3;
      u32x4 d = *(const u32x4*)(Ctx + (size_t)(rowbase + r) * N + k0 + c8 * 8);
      *(u32x4*)(&As[r * LDA + c8 * 8]) = d;
    }
#endif
    #pragma unroll
    for (int i = 0; i < 4; i++) {           // Wo tile fp32 -> bf16 transposed
      int slot = t + i * 256;
      int kr = slot >> 5, c4 = slot & 31;
      const float* wp = Wo + (size_t)(k0 + kr) * Dx + ncb + c4 * 4;
      if (k0 + 32 < N) __builtin_prefetch(wp + (size_t)32 * Dx, 0, 0);
      const f32x4 w4 = *(const f32x4*)wp;
      int cb = c4 * 4;
      Bt[(cb + 0) * LDB + kr] = (bf16)w4.x;
      Bt[(cb + 1) * LDB + kr] = (bf16)w4.y;
      Bt[(cb + 2) * LDB + kr] = (bf16)w4.z;
      Bt[(cb + 3) * LDB + kr] = (bf16)w4.w;
    }
#if HAVE_ASYNC_LDS
    __builtin_amdgcn_s_wait_asynccnt(0);
#endif
    __syncthreads();

    Frag a[4], b[2];
    #pragma unroll
    for (int tm = 0; tm < 4; tm++) {
      int row = wm * 64 + tm * 16 + l16;
      int kb0 = lh * 8;
      a[tm].q[0] = *(const u32x4*)(&As[row * LDA + kb0]);
      a[tm].q[1] = *(const u32x4*)(&As[row * LDA + kb0 + 16]);
    }
    #pragma unroll
    for (int tn = 0; tn < 2; tn++) {
      int col = wn * 32 + tn * 16 + l16;
      int kb  = lh * 8;
      b[tn].q[0] = *(const u32x4*)(&Bt[col * LDB + kb]);
      b[tn].q[1] = *(const u32x4*)(&Bt[col * LDB + kb + 16]);
    }
    #pragma unroll
    for (int tm = 0; tm < 4; tm++)
      #pragma unroll
      for (int tn = 0; tn < 2; tn++)
        acc[tm][tn] = WMMA_BF16(a[tm].v, b[tn].v, acc[tm][tn]);
  }

  #pragma unroll
  for (int tm = 0; tm < 4; tm++)
    #pragma unroll
    for (int tn = 0; tn < 2; tn++)
      #pragma unroll
      for (int r = 0; r < 8; r++) {
        int rt = rowbase + wm * 64 + tm * 16 + r + lh * 8;
        int ct = ncb + wn * 32 + tn * 16 + l16;
        Out[(size_t)rt * Dx + ct] = acc[tm][tn][r];
      }
}

// ---------------------------------------------------------------------------
extern "C" void kernel_launch(void* const* d_in, const int* in_sizes, int n_in,
                              void* d_out, int out_size, void* d_ws, size_t ws_size,
                              hipStream_t stream)
{
  (void)in_sizes; (void)n_in; (void)out_size; (void)ws_size;
  const float* X   = (const float*)d_in[0];
  // d_in[1] = positions (arange(S)); position == sequence index, used directly.
  const float* Wq  = (const float*)d_in[2];
  const float* Wk  = (const float*)d_in[3];
  const float* Wv  = (const float*)d_in[4];
  const float* Wo  = (const float*)d_in[5];
  const float* qnw = (const float*)d_in[6];
  const float* knw = (const float*)d_in[7];
  float* Out = (float*)d_out;

  // workspace carve-out (~48 MB total)
  char* ws = (char*)d_ws;
  bf16* Qb  = (bf16*)ws; ws += (size_t)Bx * HQx * Sx * HDx * sizeof(bf16);
  bf16* Kb  = (bf16*)ws; ws += (size_t)Bx * HKx * Sx * HDx * sizeof(bf16);
  bf16* Vb  = (bf16*)ws; ws += (size_t)Bx * HKx * Sx * HDx * sizeof(bf16);
  bf16* Ctx = (bf16*)ws;

  qkv_kernel <<<dim3(32, 32), 256, 0, stream>>>(X, Wq, Wk, Wv, qnw, knw, Qb, Kb, Vb);
  attn_kernel<<<dim3(Bx * HQx, Sx / 128), 256, 0, stream>>>(Qb, Kb, Vb, Ctx);
  oproj_kernel<<<dim3(16, 32), 256, 0, stream>>>(Ctx, Wo, Out);
}
